// TCRAggregator_87290915324577
// MI455X (gfx1250) — matrix operand order repaired and checked by reading, not compiled, not run on hardware
//
#include <hip/hip_runtime.h>
#include <math.h>

typedef __attribute__((ext_vector_type(16))) __bf16 v16bf;
typedef __attribute__((ext_vector_type(8)))  float  v8f;

#define D_ 128
#define H_ 64
#define L_ 256
#define BK_ 256   // B-chunk (K-span) per staging step in pool kernel
#define CH_ 4     // chunks per block: ~196 blocks, 4x fewer output atomics
#define FP_ 264   // padded LDS pitch (BK_+8): bank-conflict-free A rows

#define NEG_HUGE (-3.402823466e38f)

// branch-free tanh: 1 - 2/(exp(2x)+1); safe at +-inf
__device__ __forceinline__ float tanh_fast(float x) {
  float e = __expf(2.f * x);
  return 1.f - 2.f * __builtin_amdgcn_rcpf(e + 1.f);
}

// ---- fragment load helpers -------------------------------------------------

__device__ __forceinline__ v16bf load_frag_lds(const __bf16* p0, const __bf16* p1) {
  union { uint4 q[2]; v16bf v; } u;
  u.q[0] = *(const uint4*)p0;
  u.q[1] = *(const uint4*)p1;
  return u.v;
}

__device__ __forceinline__ v16bf load_a_frag_f32(const float* p) {
  float4 f0 = ((const float4*)p)[0];
  float4 f1 = ((const float4*)p)[1];
  float4 f2 = ((const float4*)(p + 16))[0];
  float4 f3 = ((const float4*)(p + 16))[1];
  v16bf a;
  a[0]=(__bf16)f0.x;  a[1]=(__bf16)f0.y;  a[2]=(__bf16)f0.z;  a[3]=(__bf16)f0.w;
  a[4]=(__bf16)f1.x;  a[5]=(__bf16)f1.y;  a[6]=(__bf16)f1.z;  a[7]=(__bf16)f1.w;
  a[8]=(__bf16)f2.x;  a[9]=(__bf16)f2.y;  a[10]=(__bf16)f2.z; a[11]=(__bf16)f2.w;
  a[12]=(__bf16)f3.x; a[13]=(__bf16)f3.y; a[14]=(__bf16)f3.z; a[15]=(__bf16)f3.w;
  return a;
}

// ---- kernel 0: zero output -------------------------------------------------

__global__ void zero_kernel(float* p, int n) {
  int i = blockIdx.x * blockDim.x + threadIdx.x;
  if (i < n) p[i] = 0.f;
}

// ---- kernel 1: attention scores via bf16 WMMA (+ per-block score max) ------

__global__ __launch_bounds__(256) void scores_kernel(
    const float* __restrict__ feat, const float* __restrict__ w1,
    const float* __restrict__ b1,   const float* __restrict__ w2,
    const float* __restrict__ b2,   float* __restrict__ scores,
    float* __restrict__ blockmax, int B) {
  __shared__ __align__(16) __bf16 w1t[H_ * 136];  // transposed: w1t[h][k]
  __shared__ float b1s[H_], w2s[H_];
  __shared__ float smax[16];

  for (int idx = threadIdx.x; idx < D_ * H_; idx += 256) {
    int h = idx & (H_ - 1);
    int k = idx >> 6;
    w1t[h * 136 + k] = (__bf16)w1[idx];  // w1 is [D][H] row-major
  }
  if (threadIdx.x < H_) {
    b1s[threadIdx.x] = b1[threadIdx.x];
    w2s[threadIdx.x] = w2[threadIdx.x];
  }
  __syncthreads();

  const int wave = threadIdx.x >> 5;
  const int lane = threadIdx.x & 31;
  const int half = lane >> 4;
  const int rowbase = blockIdx.x * 128 + wave * 16;
  const int row = rowbase + (lane & 15);
  const int rowc = row < B ? row : B - 1;   // clamp: garbage rows are excluded below
  const int akoff = half ? 8 : 0;
  const int bkoff = half ? 8 : 0;
  const int col = lane & 15;

  v16bf afr[4];
  #pragma unroll
  for (int c = 0; c < 4; ++c)
    afr[c] = load_a_frag_f32(feat + (size_t)rowc * D_ + c * 32 + akoff);

  float part[8] = {0.f,0.f,0.f,0.f,0.f,0.f,0.f,0.f};
  #pragma unroll
  for (int j = 0; j < 4; ++j) {
    v16bf bfr[4];
    #pragma unroll
    for (int c = 0; c < 4; ++c) {
      const __bf16* bp = &w1t[(16 * j + col) * 136 + c * 32 + bkoff];
      bfr[c] = load_frag_lds(bp, bp + 16);
    }
    v8f acc = {};
    #pragma unroll
    for (int c = 0; c < 4; ++c)
      acc = __builtin_amdgcn_wmma_f32_16x16x32_bf16(
          false, afr[c], false, bfr[c], (short)0, acc, false, false);

    float b1v = b1s[16 * j + col];
    float w2v = w2s[16 * j + col];
    #pragma unroll
    for (int r = 0; r < 8; ++r) part[r] += tanh_fast(acc[r] + b1v) * w2v;
  }

  #pragma unroll
  for (int r = 0; r < 8; ++r) {
    float v = part[r];
    v += __shfl_xor(v, 8, 32);
    v += __shfl_xor(v, 4, 32);
    v += __shfl_xor(v, 2, 32);
    v += __shfl_xor(v, 1, 32);
    part[r] = v;
  }
  if ((lane & 15) == 0) {
    const float b2v = b2[0];
    const int mbase = half << 3;
    float lmax = NEG_HUGE;
    #pragma unroll
    for (int r = 0; r < 8; ++r) {
      int row2 = rowbase + mbase + r;
      float sc = part[r] + b2v;
      if (row2 < B) { scores[row2] = sc; lmax = fmaxf(lmax, sc); }
    }
    smax[wave * 2 + half] = lmax;
  }
  __syncthreads();
  if (threadIdx.x == 0) {
    float m = smax[0];
    #pragma unroll
    for (int i = 1; i < 16; ++i) m = fmaxf(m, smax[i]);
    blockmax[blockIdx.x] = m;
  }
}

// ---- kernel 1b: reduce block maxes to a single global max ------------------

__global__ __launch_bounds__(256) void gmax_kernel(
    const float* __restrict__ part, int n, float* __restrict__ gmax) {
  float m = NEG_HUGE;
  for (int i = threadIdx.x; i < n; i += 256) m = fmaxf(m, part[i]);
  __shared__ float rm[256];
  rm[threadIdx.x] = m;
  __syncthreads();
  for (int off = 128; off > 0; off >>= 1) {
    if ((int)threadIdx.x < off)
      rm[threadIdx.x] = fmaxf(rm[threadIdx.x], rm[threadIdx.x + off]);
    __syncthreads();
  }
  if (threadIdx.x == 0) gmax[0] = rm[0];
}

// ---- kernel 1c: escore[b] = exp(score[b] - gmax) (one exp per sequence) ----

__global__ __launch_bounds__(256) void escore_kernel(
    const float* __restrict__ scores, const float* __restrict__ gmax,
    float* __restrict__ escore, int B) {
  int i = blockIdx.x * blockDim.x + threadIdx.x;
  if (i < B) escore[i] = __expf(scores[i] - gmax[0]);
}

// ---- kernel 2: per-library masked sum of escore (pure streaming) -----------
// softmax shift-invariance: w[l,b] = escore[b] / sum_members(escore)

__global__ __launch_bounds__(256) void libsum_kernel(
    const int* __restrict__ mask, const float* __restrict__ escore,
    float* __restrict__ invS, int B) {
  const int l = blockIdx.x;
  const int* mrow = mask + (size_t)l * B;
  float s = 0.f;
  const int B4 = B >> 2;
  for (int q = threadIdx.x; q < B4; q += blockDim.x) {
    int4   mk = ((const int4*)mrow)[q];
    float4 e  = ((const float4*)escore)[q];
    s += (mk.x > 0) ? e.x : 0.f;
    s += (mk.y > 0) ? e.y : 0.f;
    s += (mk.z > 0) ? e.z : 0.f;
    s += (mk.w > 0) ? e.w : 0.f;
  }
  for (int b = (B4 << 2) + threadIdx.x; b < B; b += blockDim.x)
    s += (mrow[b] > 0) ? escore[b] : 0.f;

  __shared__ float rs[256];
  rs[threadIdx.x] = s;
  __syncthreads();
  for (int off = 128; off > 0; off >>= 1) {
    if ((int)threadIdx.x < off) rs[threadIdx.x] += rs[threadIdx.x + off];
    __syncthreads();
  }
  if (threadIdx.x == 0) invS[l] = rs[0] > 0.f ? 1.f / rs[0] : 0.f;  // empty -> 0
}

// ---- kernel 3: pooled output via bf16 WMMA ---------------------------------
// Each block owns CH_ B-chunks; two passes of 8 library tiles with persistent
// register accumulators -> one atomic flush per block. Weights are a single
// mul+select: w = member ? escore_b * invS_l : 0.

__global__ __launch_bounds__(256) void pool_kernel(
    const float* __restrict__ feat, const int* __restrict__ mask,
    const float* __restrict__ escore, const float* __restrict__ invS,
    float* __restrict__ out, int B) {
  __shared__ __align__(16) __bf16 featT[D_ * FP_];   // featT[d][b], bf16
  __shared__ __align__(16) __bf16 wrow[16 * FP_];    // weights tile [m][b]
  __shared__ float s_sc[BK_], s_iz[L_];

  const int tid = threadIdx.x;
  const int wave = tid >> 5;
  const int lane = tid & 31;
  const int half = lane >> 4;
  const int dcol = wave * 16 + (lane & 15);
  const int cg = tid & 63;                   // column group (4 b's)
  const int rg = tid >> 6;                   // row group (4 m's)

  s_iz[tid] = invS[tid];   // L_ == blockDim == 256

  const int c0 = blockIdx.x * CH_;

  for (int pass = 0; pass < 2; ++pass) {
    v8f acc[8];
    #pragma unroll
    for (int j = 0; j < 8; ++j) acc[j] = (v8f){};

    for (int cc = 0; cc < CH_; ++cc) {
      const int chunk = c0 + cc;
      const int b0 = chunk * BK_;
      if (b0 >= B) break;                    // uniform per block

      __syncthreads();
      for (int idx = tid; idx < (BK_ * D_) / 4; idx += 256) {
        int lin = idx << 2;
        int d = lin & (D_ - 1);
        int b = lin >> 7;
        int gb = (b0 + b) < B ? (b0 + b) : (B - 1);
        float4 f = *(const float4*)&feat[(size_t)gb * D_ + d];
        featT[(d + 0) * FP_ + b] = (__bf16)f.x;
        featT[(d + 1) * FP_ + b] = (__bf16)f.y;
        featT[(d + 2) * FP_ + b] = (__bf16)f.z;
        featT[(d + 3) * FP_ + b] = (__bf16)f.w;
      }
      s_sc[tid] = escore[(b0 + tid) < B ? (b0 + tid) : (B - 1)];
      __syncthreads();

      const int bbase = b0 + cg * 4;
      const int bclamp = bbase <= (B - 4) ? bbase : (B - 4);
      const float4 sc4 = *(const float4*)&s_sc[cg * 4];

      #pragma unroll
      for (int j = 0; j < 8; ++j) {
        const int lt = pass * 8 + j;
        #pragma unroll
        for (int mm = 0; mm < 4; ++mm) {
          int m = rg * 4 + mm;
          int l = lt * 16 + m;
          const int* mp = &mask[(size_t)l * B + bclamp];
          int4 mk = *(const int4*)mp;
          if (cc + 1 < CH_ && b0 + BK_ < B)
            __builtin_prefetch(mp + BK_, 0, 1);   // next chunk, same rows
          float izl = s_iz[l];
          union { __bf16 h[4]; uint2 q; } pk;
          pk.h[0] = (__bf16)(((bbase + 0) < B && mk.x > 0) ? sc4.x * izl : 0.f);
          pk.h[1] = (__bf16)(((bbase + 1) < B && mk.y > 0) ? sc4.y * izl : 0.f);
          pk.h[2] = (__bf16)(((bbase + 2) < B && mk.z > 0) ? sc4.z * izl : 0.f);
          pk.h[3] = (__bf16)(((bbase + 3) < B && mk.w > 0) ? sc4.w * izl : 0.f);
          *(uint2*)&wrow[m * FP_ + cg * 4] = pk.q;
        }
        __syncthreads();

        const __bf16* arow = &wrow[(lane & 15) * FP_ + (half ? 8 : 0)];
        const __bf16* brow = &featT[dcol * FP_ + (half ? 8 : 0)];
        v16bf a0 = load_frag_lds(arow, arow + 16);
        v16bf bb0 = load_frag_lds(brow, brow + 16);
        #pragma unroll
        for (int kc = 0; kc < BK_ / 32; ++kc) {
          v16bf a1, bb1;
          if (kc + 1 < BK_ / 32) {
            a1  = load_frag_lds(arow + (kc + 1) * 32, arow + (kc + 1) * 32 + 16);
            bb1 = load_frag_lds(brow + (kc + 1) * 32, brow + (kc + 1) * 32 + 16);
          }
          acc[j] = __builtin_amdgcn_wmma_f32_16x16x32_bf16(
              false, a0, false, bb0, (short)0, acc[j], false, false);
          a0 = a1; bb0 = bb1;
        }
        __syncthreads();
      }
    }

    #pragma unroll
    for (int j = 0; j < 8; ++j) {
      #pragma unroll
      for (int r = 0; r < 8; ++r) {
        int l = (pass * 8 + j) * 16 + r + (half << 3);
        unsafeAtomicAdd(&out[l * D_ + dcol], acc[j][r]);
      }
    }
  }
}

// ---- launch ----------------------------------------------------------------

extern "C" void kernel_launch(void* const* d_in, const int* in_sizes, int n_in,
                              void* d_out, int out_size, void* d_ws, size_t ws_size,
                              hipStream_t stream) {
  const float* feat = (const float*)d_in[0];
  const int*   mask = (const int*)d_in[1];
  const float* w1   = (const float*)d_in[2];
  const float* b1   = (const float*)d_in[3];
  const float* w2   = (const float*)d_in[4];
  const float* b2   = (const float*)d_in[5];
  float* out = (float*)d_out;

  const int B = in_sizes[0] / D_;
  const int L = in_sizes[1] / B;   // 256

  const int nsb = (B + 127) / 128;          // scores blocks
  float* wsf      = (float*)d_ws;
  float* scores   = wsf;                    // B floats
  float* escore   = scores + B;             // B floats
  float* invS     = escore + B;             // L floats
  float* blockmax = invS + L;               // nsb floats
  float* gmax     = blockmax + nsb;         // 1 float

  const int nchunks = (B + BK_ - 1) / BK_;
  const int nblocks = (nchunks + CH_ - 1) / CH_;

  zero_kernel<<<(out_size + 255) / 256, 256, 0, stream>>>(out, out_size);
  scores_kernel<<<nsb, 256, 0, stream>>>(feat, w1, b1, w2, b2, scores, blockmax, B);
  gmax_kernel<<<1, 256, 0, stream>>>(blockmax, nsb, gmax);
  escore_kernel<<<(B + 255) / 256, 256, 0, stream>>>(scores, gmax, escore, B);
  libsum_kernel<<<L, 256, 0, stream>>>(mask, escore, invS, B);
  pool_kernel<<<nblocks, 256, 0, stream>>>(feat, mask, escore, invS, out, B);
}
